// GAT_26087631356165
// MI455X (gfx1250) — matrix operand order repaired and checked by reading, not compiled, not run on hardware
//
#include <hip/hip_runtime.h>
#include <hip/hip_bf16.h>
#include <math.h>

typedef __attribute__((ext_vector_type(16))) _Float16 v16h;
typedef __attribute__((ext_vector_type(8)))  _Float16 v8h;
typedef __attribute__((ext_vector_type(4)))  _Float16 v4h;
typedef __attribute__((ext_vector_type(8)))  float    v8f;

#define TPB 256
#define BM 64
#define BN 32
#define BK 32
#define LDA  40   // padded f16 stride (A tile rows), 80B => 16B aligned rows
#define LDBT 40   // padded f16 stride (B^T tile rows)
#define NEG_SLOPE 0.2f

static inline int cdiv(int a, int b) { return (a + b - 1) / b; }

// ---------------------------------------------------------------------------
// Tiled GEMM C[M,N] = A[M,K] @ W[K,N], f32 in -> f16 fragments -> f32 acc.
// 256 thr = 8 waves; wave (wm,wn) owns a 16x16 C tile of a 64x32 block tile.
// K is a compile-time constant (128 for both layers) -> fully unrolled,
// 4 v_wmma per pass, fragments loaded as aligned ds_load_b128 pairs.
// ---------------------------------------------------------------------------
template <int K>
__global__ __launch_bounds__(TPB)
void gat_gemm_wmma(const float* __restrict__ A, const float* __restrict__ W,
                   float* __restrict__ C, int M, int N) {
  __shared__ __align__(16) _Float16 As[BM * LDA];   // [row][k]
  __shared__ __align__(16) _Float16 Bs[BN * LDBT];  // [col][k]  (transposed)

  const int tid  = threadIdx.x;
  const int lane = tid & 31;
  const int wave = tid >> 5;      // 0..7
  const int wm   = wave & 3;      // row sub-tile
  const int wn   = wave >> 2;     // col sub-tile
  const int block_row = blockIdx.x * BM;
  const int block_col = blockIdx.y * BN;
  const int half16 = lane >> 4;   // lane group 0/1
  const int l15    = lane & 15;

  // Static staging slices: A -> thread owns 8 contiguous K of one row;
  // B -> thread owns 4 contiguous K of one column (stored transposed).
  const int ar = tid >> 2;              // 0..63  A row in tile
  const int ac = (tid & 3) * 8;         // 0,8,16,24
  const int bc = tid >> 3;              // 0..31  B col in tile
  const int bk = (tid & 7) * 4;         // 0..28
  const bool arow_ok = (block_row + ar) < M;
  const float* __restrict__ Arow = A + (size_t)(block_row + ar) * K;
  const float* __restrict__ Wcol = W + (size_t)bk * N + block_col + bc;

  v8f acc = {};

#pragma unroll
  for (int k0 = 0; k0 < K; k0 += BK) {
    // ---- stage A (two float4 loads -> one 16B LDS store) ----
    float4 a0 = {0.f, 0.f, 0.f, 0.f}, a1 = {0.f, 0.f, 0.f, 0.f};
    if (arow_ok) {
      a0 = *(const float4*)(Arow + k0 + ac);
      a1 = *(const float4*)(Arow + k0 + ac + 4);
    }
    v8h a8;
    a8[0] = (_Float16)a0.x; a8[1] = (_Float16)a0.y;
    a8[2] = (_Float16)a0.z; a8[3] = (_Float16)a0.w;
    a8[4] = (_Float16)a1.x; a8[5] = (_Float16)a1.y;
    a8[6] = (_Float16)a1.z; a8[7] = (_Float16)a1.w;
    *(v8h*)&As[ar * LDA + ac] = a8;

    // ---- stage B transposed (4 strided loads -> one 8B LDS store) ----
    const float* wp = Wcol + (size_t)k0 * N;
    v4h b4;
    b4[0] = (_Float16)wp[0];
    b4[1] = (_Float16)wp[N];
    b4[2] = (_Float16)wp[2 * N];
    b4[3] = (_Float16)wp[3 * N];
    *(v4h*)&Bs[bc * LDBT + bk] = b4;

    __syncthreads();

    // ---- A fragment: lanes 0-15 K={0..7,16..23}, lanes 16-31 +8 ----
    const _Float16* ap = &As[(wm * 16 + l15) * LDA + half16 * 8];
    v8h alo = *(const v8h*)ap;          // K = h*8 .. h*8+7
    v8h ahi = *(const v8h*)(ap + 16);   // K = 16+h*8 .. +7
    v16h af = __builtin_shufflevector(alo, ahi,
        0, 1, 2, 3, 4, 5, 6, 7, 8, 9, 10, 11, 12, 13, 14, 15);

    // ---- B fragment: lanes 0-15 K={0..7,16..23}, lanes 16-31 +8 ----
    const _Float16* bp = &Bs[(wn * 16 + l15) * LDBT + half16 * 8];
    v8h blo = *(const v8h*)bp;
    v8h bhi = *(const v8h*)(bp + 16);
    v16h bf = __builtin_shufflevector(blo, bhi,
        0, 1, 2, 3, 4, 5, 6, 7, 8, 9, 10, 11, 12, 13, 14, 15);

    acc = __builtin_amdgcn_wmma_f32_16x16x32_f16(
        /*neg_a=*/false, af, /*neg_b=*/false, bf,
        /*c_mod=*/(short)0, acc, /*reuse_a=*/false, /*reuse_b=*/false);
    __syncthreads();
  }

  // C/D layout: VGPR r holds M=r (lanes 0-15) / M=r+8 (lanes 16-31), N=l15
  const int col   = block_col + wn * 16 + l15;
  const int rbase = block_row + wm * 16 + (half16 << 3);
  if (col < N) {
#pragma unroll
    for (int r = 0; r < 8; ++r) {
      int row = rbase + r;
      if (row < M) C[(size_t)row * N + col] = acc[r];
    }
  }
}

// ---------------------------------------------------------------------------
// Attention scores: s_src[n,h] = <h[n,h,:], a_src[h,:]>, same for dst.
// ---------------------------------------------------------------------------
__global__ void gat_scores(const float* __restrict__ h,
                           const float* __restrict__ a_src,
                           const float* __restrict__ a_dst,
                           float* __restrict__ s_src, float* __restrict__ s_dst,
                           int N, int H, int Cc) {
  int idx = blockIdx.x * blockDim.x + threadIdx.x;
  if (idx >= N * H) return;
  int n = idx / H, hh = idx - n * H;
  const float* row = h + (size_t)n * H * Cc + (size_t)hh * Cc;
  const float* as  = a_src + hh * Cc;
  const float* ad  = a_dst + hh * Cc;
  float ss = 0.f, sd = 0.f;
  for (int c = 0; c < Cc; c += 4) {
    float4 v = *(const float4*)(row + c);
    float4 s4 = *(const float4*)(as + c);
    float4 d4 = *(const float4*)(ad + c);
    ss += v.x * s4.x + v.y * s4.y + v.z * s4.z + v.w * s4.w;
    sd += v.x * d4.x + v.y * d4.y + v.z * d4.z + v.w * d4.w;
  }
  s_src[idx] = ss;
  s_dst[idx] = sd;
}

__global__ void gat_fill(float* __restrict__ p, float v, int n) {
  int i = blockIdx.x * blockDim.x + threadIdx.x;
  if (i < n) p[i] = v;
}

// Monotonic-int-encoding float atomic max (pure integer atomics).
__device__ inline void atomicMaxF(float* addr, float value) {
  if (value >= 0.f)
    atomicMax((int*)addr, __float_as_int(value));
  else
    atomicMin((unsigned int*)addr, __float_as_uint(value));
}

__device__ inline void edge_pair(const int* __restrict__ ei, int e, int E,
                                 int& s, int& d) {
  if (e < E) { s = ei[e]; d = ei[E + e]; }
  else       { s = e - E; d = e - E; }     // synthesized self-loop
}

// Pass 1: segment max of leaky-relu edge logits.
__global__ void gat_edge_max(const int* __restrict__ ei,
                             const float* __restrict__ s_src,
                             const float* __restrict__ s_dst,
                             float* __restrict__ emax, int E, int N, int H) {
  int idx = blockIdx.x * blockDim.x + threadIdx.x;
  int tot = (E + N) * H;
  if (idx >= tot) return;
  int e = idx / H, hh = idx - e * H;
  int s, d; edge_pair(ei, e, E, s, d);
  float x = s_src[s * H + hh] + s_dst[d * H + hh];
  x = x > 0.f ? x : NEG_SLOPE * x;
  atomicMaxF(&emax[d * H + hh], x);
}

// Pass 2: segment sum of exp(e - emax).
__global__ void gat_edge_sum(const int* __restrict__ ei,
                             const float* __restrict__ s_src,
                             const float* __restrict__ s_dst,
                             const float* __restrict__ emax,
                             float* __restrict__ den, int E, int N, int H) {
  int idx = blockIdx.x * blockDim.x + threadIdx.x;
  int tot = (E + N) * H;
  if (idx >= tot) return;
  int e = idx / H, hh = idx - e * H;
  int s, d; edge_pair(ei, e, E, s, d);
  float x = s_src[s * H + hh] + s_dst[d * H + hh];
  x = x > 0.f ? x : NEG_SLOPE * x;
  atomicAdd(&den[d * H + hh], __expf(x - emax[d * H + hh]));
}

// Pass 3: out[dst] += alpha * h[src]. One wave per edge, HC/32 channels/lane.
__global__ __launch_bounds__(TPB)
void gat_edge_aggregate(const int* __restrict__ ei,
                        const float* __restrict__ s_src,
                        const float* __restrict__ s_dst,
                        const float* __restrict__ emax,
                        const float* __restrict__ den,
                        const float* __restrict__ hfeat,
                        float* __restrict__ out,
                        int E, int N, int H, int Cc) {
  const int HC   = H * Cc;
  const int edge = blockIdx.x * (TPB / 32) + (threadIdx.x >> 5);
  const int lane = threadIdx.x & 31;
  if (edge >= E + N) return;
  int s, d; edge_pair(ei, edge, E, s, d);

  const int cpl   = HC >> 5;         // 4 (layer1) or 1 (layer2)
  const int cbase = lane * cpl;
  const int head  = cbase / Cc;      // cpl divides Cc -> one head per lane

  const float* hrow = hfeat + (size_t)s * HC + cbase;
  __builtin_prefetch(hrow, 0, 0);

  float x = s_src[s * H + head] + s_dst[d * H + head];
  x = x > 0.f ? x : NEG_SLOPE * x;
  const float alpha = __expf(x - emax[d * H + head]) / den[d * H + head];

  float* orow = out + (size_t)d * HC + cbase;
  if (cpl == 4) {
    float4 v = *(const float4*)hrow;
    atomicAdd(&orow[0], v.x * alpha);
    atomicAdd(&orow[1], v.y * alpha);
    atomicAdd(&orow[2], v.z * alpha);
    atomicAdd(&orow[3], v.w * alpha);
  } else {
    atomicAdd(&orow[0], hrow[0] * alpha);
  }
}

__global__ void gat_bias_elu(float* __restrict__ h, const float* __restrict__ b,
                             int N, int C) {
  int i = blockIdx.x * blockDim.x + threadIdx.x;
  if (i >= N * C) return;
  float v = h[i] + b[i % C];
  h[i] = v > 0.f ? v : (__expf(v) - 1.f);
}

__global__ void gat_bias_add(float* __restrict__ o, const float* __restrict__ b,
                             int N, int C) {
  int i = blockIdx.x * blockDim.x + threadIdx.x;
  if (i >= N * C) return;
  o[i] += b[i % C];
}

// ---------------------------------------------------------------------------
extern "C" void kernel_launch(void* const* d_in, const int* in_sizes, int n_in,
                              void* d_out, int out_size, void* d_ws, size_t ws_size,
                              hipStream_t stream) {
  const float* x      = (const float*)d_in[0];
  const int*   ei     = (const int*)  d_in[1];
  const float* W1     = (const float*)d_in[2];
  const float* a1_src = (const float*)d_in[3];
  const float* a1_dst = (const float*)d_in[4];
  const float* b1     = (const float*)d_in[5];
  const float* W2     = (const float*)d_in[6];
  const float* a2_src = (const float*)d_in[7];
  const float* a2_dst = (const float*)d_in[8];
  const float* b2     = (const float*)d_in[9];
  float*       outp   = (float*)d_out;

  const int IN = 128, H1 = 2, C1 = 64, HC1 = H1 * C1, H2 = 1, C2 = 32;
  const int N = in_sizes[0] / IN;        // 50000
  const int E = in_sizes[1] / 2;         // 800000
  const int ET = E + N;                  // with self-loops

  float* ws = (float*)d_ws;
  size_t off = 0;
  float* h1   = ws + off; off += (size_t)N * HC1;   // GEMM1 output
  float* agg1 = ws + off; off += (size_t)N * HC1;   // layer-1 aggregate -> elu(h)
  float* h2   = ws + off; off += (size_t)N * C2;    // GEMM2 output
  float* s1s  = ws + off; off += (size_t)N * H1;
  float* s1d  = ws + off; off += (size_t)N * H1;
  float* e1   = ws + off; off += (size_t)N * H1;
  float* d1   = ws + off; off += (size_t)N * H1;
  float* s2s  = ws + off; off += (size_t)N * H2;
  float* s2d  = ws + off; off += (size_t)N * H2;
  float* e2   = ws + off; off += (size_t)N * H2;
  float* d2   = ws + off; off += (size_t)N * H2;

  const dim3 blk(TPB);

  // ---- Layer 1 ----
  {
    dim3 grid(cdiv(N, BM), HC1 / BN);
    gat_gemm_wmma<128><<<grid, blk, 0, stream>>>(x, W1, h1, N, HC1);
  }
  gat_scores<<<cdiv(N * H1, TPB), blk, 0, stream>>>(h1, a1_src, a1_dst, s1s, s1d, N, H1, C1);
  gat_fill<<<cdiv(N * H1, TPB), blk, 0, stream>>>(e1, -INFINITY, N * H1);
  gat_fill<<<cdiv(N * H1, TPB), blk, 0, stream>>>(d1, 0.f, N * H1);
  gat_fill<<<cdiv(N * HC1, TPB), blk, 0, stream>>>(agg1, 0.f, N * HC1);
  gat_edge_max<<<cdiv(ET * H1, TPB), blk, 0, stream>>>(ei, s1s, s1d, e1, E, N, H1);
  gat_edge_sum<<<cdiv(ET * H1, TPB), blk, 0, stream>>>(ei, s1s, s1d, e1, d1, E, N, H1);
  gat_edge_aggregate<<<cdiv(ET, TPB / 32), blk, 0, stream>>>(ei, s1s, s1d, e1, d1, h1, agg1, E, N, H1, C1);
  gat_bias_elu<<<cdiv(N * HC1, TPB), blk, 0, stream>>>(agg1, b1, N, HC1);

  // ---- Layer 2 ----
  {
    dim3 grid(cdiv(N, BM), C2 / BN);
    gat_gemm_wmma<128><<<grid, blk, 0, stream>>>(agg1, W2, h2, N, C2);
  }
  gat_scores<<<cdiv(N * H2, TPB), blk, 0, stream>>>(h2, a2_src, a2_dst, s2s, s2d, N, H2, C2);
  gat_fill<<<cdiv(N * H2, TPB), blk, 0, stream>>>(e2, -INFINITY, N * H2);
  gat_fill<<<cdiv(N * H2, TPB), blk, 0, stream>>>(d2, 0.f, N * H2);
  gat_fill<<<cdiv(N * C2, TPB), blk, 0, stream>>>(outp, 0.f, N * C2);
  gat_edge_max<<<cdiv(ET * H2, TPB), blk, 0, stream>>>(ei, s2s, s2d, e2, E, N, H2);
  gat_edge_sum<<<cdiv(ET * H2, TPB), blk, 0, stream>>>(ei, s2s, s2d, e2, d2, E, N, H2);
  gat_edge_aggregate<<<cdiv(ET, TPB / 32), blk, 0, stream>>>(ei, s2s, s2d, e2, d2, h2, outp, E, N, H2, C2);
  gat_bias_add<<<cdiv(N * C2, TPB), blk, 0, stream>>>(outp, b2, N, C2);
}
